// GlobalAttentionNet_35622458753572
// MI455X (gfx1250) — matrix operand order repaired and checked by reading, not compiled, not run on hardware
//
#include <hip/hip_runtime.h>
#include <hip/hip_bf16.h>

#define N_NODES   50000
#define N_EDGES   800000
#define N_FEAT    128
#define HIDDEN    128
#define N_CLASSES 10
#define N_GRAPHS  512

typedef __attribute__((ext_vector_type(16))) __bf16 v16bf;
typedef __attribute__((ext_vector_type(8)))  __bf16 v8bf;
typedef __attribute__((ext_vector_type(4)))  __bf16 v4bf;
typedef __attribute__((ext_vector_type(8)))  float  v8f;

union BFrag { v16bf v; v8bf h[2]; };

// ---------------------------------------------------------------- utilities
__global__ void k_zero(float* __restrict__ p, int n) {
    int i = blockIdx.x * blockDim.x + threadIdx.x;
    if (i < n) p[i] = 0.0f;
}

__global__ void k_deg(const int* __restrict__ ei, float* __restrict__ deg) {
    int e = blockIdx.x * blockDim.x + threadIdx.x;
    if (e < N_EDGES) atomicAdd(&deg[ei[N_EDGES + e]], 1.0f);
}

__global__ void k_invdeg(float* __restrict__ d) {
    int i = blockIdx.x * blockDim.x + threadIdx.x;
    if (i < N_NODES) d[i] = 1.0f / fmaxf(d[i], 1.0f);
}

// transpose + convert one 128x128 f32 [K][N] weight to bf16 [N][K]
__global__ void k_wprep(const float* __restrict__ w, __bf16* __restrict__ wt) {
    int i = blockIdx.x * blockDim.x + threadIdx.x;
    if (i < HIDDEN * HIDDEN) {
        int k = i >> 7, n = i & 127;
        wt[n * HIDDEN + k] = (__bf16)w[k * HIDDEN + n];
    }
}

// edge scatter: agg[dst] += x[src]; one wave per edge, 4 feats per lane
__global__ void k_scatter(const int* __restrict__ ei, const float* __restrict__ x,
                          float* __restrict__ agg) {
    int t = blockIdx.x * blockDim.x + threadIdx.x;
    int e = t >> 5, lane = t & 31;
    if (e >= N_EDGES) return;
    int s = ei[e];
    int d = ei[N_EDGES + e];
    const float4 v = *(const float4*)&x[s * HIDDEN + lane * 4];
    float* o = &agg[d * HIDDEN + lane * 4];
    atomicAdd(o + 0, v.x);
    atomicAdd(o + 1, v.y);
    atomicAdd(o + 2, v.z);
    atomicAdd(o + 3, v.w);
}

// ------------------------------------------------ SAGE GEMM (WMMA bf16->f32)
// out = relu( (agg*inv_deg) @ Wl + x @ Wr + b ), weights pre-transposed bf16 [N][K]
// 128 threads = 4 waves; each wave: 16 rows x 128 cols; block: 64 rows.
#define LDSPAD 136
__global__ __launch_bounds__(128) void k_sage_gemm(
        const float* __restrict__ aggS, const float* __restrict__ xS,
        const float* __restrict__ invDeg,
        const __bf16* __restrict__ wlT, const __bf16* __restrict__ wrT,
        const float* __restrict__ bias, float* __restrict__ out) {
    __shared__ __bf16 sA[64][LDSPAD];
    __shared__ __bf16 sW[128][LDSPAD];

    const int tid  = threadIdx.x;
    const int wave = tid >> 5;
    const int lane = tid & 31;
    const int rowBase = blockIdx.x * 64;

    v8f acc[8] = {};

    for (int pass = 0; pass < 2; ++pass) {
        const float*  A  = pass ? xS  : aggS;
        const __bf16* Wt = pass ? wrT : wlT;
        __syncthreads();
        // stage A tile: 64x128 f32 -> bf16 (scaled by inv_deg on pass 0)
        #pragma unroll
        for (int i = 0; i < 16; ++i) {
            int idx = tid + i * 128;           // 0..2047 float4 chunks
            int r = idx >> 5, c4 = idx & 31;
            int grow = rowBase + r;
            float4 v = {0.f, 0.f, 0.f, 0.f};
            float sc = 1.0f;
            if (grow < N_NODES) {
                v = *(const float4*)&A[grow * HIDDEN + c4 * 4];
                if (!pass) {
                    sc = invDeg[grow];
                    __builtin_prefetch(&xS[grow * HIDDEN + c4 * 4], 0, 0);
                }
            }
            v4bf b;
            b[0] = (__bf16)(v.x * sc); b[1] = (__bf16)(v.y * sc);
            b[2] = (__bf16)(v.z * sc); b[3] = (__bf16)(v.w * sc);
            *(v4bf*)&sA[r][c4 * 4] = b;
        }
        // stage W^T tile: 128x128 bf16
        #pragma unroll
        for (int i = 0; i < 32; ++i) {
            int idx = tid + i * 128;           // 0..4095 v4bf chunks
            int r = idx >> 5, c4 = idx & 31;
            *(v4bf*)&sW[r][c4 * 4] = *(const v4bf*)&Wt[r * HIDDEN + c4 * 4];
        }
        __syncthreads();

        const int mrow = lane & 15;            // M for A, N for B/D
        const int kbA  = (lane >> 4) * 16;
        const int kbB  = (lane >> 4) * 16;
        #pragma unroll
        for (int kk = 0; kk < HIDDEN; kk += 32) {
            BFrag fa;
            fa.h[0] = *(const v8bf*)&sA[wave * 16 + mrow][kk + kbA];
            fa.h[1] = *(const v8bf*)&sA[wave * 16 + mrow][kk + kbA + 8];
            #pragma unroll
            for (int ct = 0; ct < 8; ++ct) {
                BFrag fb;
                fb.h[0] = *(const v8bf*)&sW[ct * 16 + mrow][kk + kbB];
                fb.h[1] = *(const v8bf*)&sW[ct * 16 + mrow][kk + kbB + 8];
                acc[ct] = __builtin_amdgcn_wmma_f32_16x16x32_bf16(
                    false, fa.v, false, fb.v, (short)0, acc[ct], false, false);
            }
        }
    }

    // bias + relu + store: D layout M = v + 8*(lane>>4), N = lane&15
    const int ncol = lane & 15;
    const int vOff = (lane >> 4) * 8;
    #pragma unroll
    for (int ct = 0; ct < 8; ++ct) {
        int col = ct * 16 + ncol;
        float b = bias[col];
        #pragma unroll
        for (int v = 0; v < 8; ++v) {
            int grow = rowBase + wave * 16 + vOff + v;
            if (grow < N_NODES)
                out[grow * HIDDEN + col] = fmaxf(acc[ct][v] + b, 0.0f);
        }
    }
}

// ------------------------------------------------------- attention pooling
// gate[i] = h[i,:] . gate_w + gate_b ; segment max via order-preserving uint
__global__ void k_gate(const float* __restrict__ h, const float* __restrict__ gw,
                       const float* __restrict__ gb, const int* __restrict__ batch,
                       float* __restrict__ gate, unsigned* __restrict__ mEnc) {
    int node = blockIdx.x * (blockDim.x >> 5) + (threadIdx.x >> 5);
    int lane = threadIdx.x & 31;
    if (node >= N_NODES) return;
    const float4 hv = *(const float4*)&h[node * HIDDEN + lane * 4];
    const float4 gv = *(const float4*)&gw[lane * 4];
    float s = hv.x * gv.x + hv.y * gv.y + hv.z * gv.z + hv.w * gv.w;
    for (int off = 16; off > 0; off >>= 1) s += __shfl_down(s, off, 32);
    if (lane == 0) {
        s += gb[0];
        gate[node] = s;
        unsigned u = __float_as_uint(s);
        unsigned enc = (s >= 0.0f) ? (u | 0x80000000u) : ~u;
        atomicMax(&mEnc[batch[node]], enc);
    }
}

__global__ void k_mdec(const unsigned* __restrict__ mEnc, float* __restrict__ m) {
    int g = blockIdx.x * blockDim.x + threadIdx.x;
    if (g >= N_GRAPHS) return;
    unsigned e = mEnc[g];
    float val = 0.0f;
    if (e != 0u) {
        unsigned u = (e & 0x80000000u) ? (e & 0x7FFFFFFFu) : ~e;
        val = __uint_as_float(u);
        if (!__builtin_isfinite(val)) val = 0.0f;
    }
    m[g] = val;
}

__global__ void k_exp(const float* __restrict__ gate, const int* __restrict__ batch,
                      const float* __restrict__ m, float* __restrict__ g,
                      float* __restrict__ den) {
    int i = blockIdx.x * blockDim.x + threadIdx.x;
    if (i >= N_NODES) return;
    int b = batch[i];
    float v = expf(gate[i] - m[b]);
    g[i] = v;
    atomicAdd(&den[b], v);
}

__global__ void k_pool(const float* __restrict__ h, const float* __restrict__ g,
                       const float* __restrict__ den, const int* __restrict__ batch,
                       float* __restrict__ pooled) {
    int t = blockIdx.x * blockDim.x + threadIdx.x;
    int node = t >> 5, lane = t & 31;
    if (node >= N_NODES) return;
    int b = batch[node];
    float a = g[node] / den[b];
    const float4 hv = *(const float4*)&h[node * HIDDEN + lane * 4];
    float* o = &pooled[b * HIDDEN + lane * 4];
    atomicAdd(o + 0, a * hv.x);
    atomicAdd(o + 1, a * hv.y);
    atomicAdd(o + 2, a * hv.z);
    atomicAdd(o + 3, a * hv.w);
}

// --------------------------------------------------------------- MLP head
__global__ __launch_bounds__(128) void k_lin1(const float* __restrict__ pooled,
                                              const float* __restrict__ w,
                                              const float* __restrict__ b,
                                              float* __restrict__ out) {
    __shared__ float row[HIDDEN];
    int g = blockIdx.x, j = threadIdx.x;
    row[j] = pooled[g * HIDDEN + j];
    __syncthreads();
    float s = b[j];
    #pragma unroll 8
    for (int k = 0; k < HIDDEN; ++k) s += row[k] * w[k * HIDDEN + j];
    out[g * HIDDEN + j] = fmaxf(s, 0.0f);
}

__global__ __launch_bounds__(32) void k_lin2(const float* __restrict__ x,
                                             const float* __restrict__ w,
                                             const float* __restrict__ b,
                                             float* __restrict__ out) {
    __shared__ float row[HIDDEN];
    __shared__ float logits[N_CLASSES];
    __shared__ float red[2];
    int g = blockIdx.x, t = threadIdx.x;
    for (int i = t; i < HIDDEN; i += 32) row[i] = x[g * HIDDEN + i];
    __syncthreads();
    if (t < N_CLASSES) {
        float s = b[t];
        for (int k = 0; k < HIDDEN; ++k) s += row[k] * w[k * N_CLASSES + t];
        logits[t] = s;
    }
    __syncthreads();
    if (t == 0) {
        float mx = logits[0];
        for (int c = 1; c < N_CLASSES; ++c) mx = fmaxf(mx, logits[c]);
        float sum = 0.0f;
        for (int c = 0; c < N_CLASSES; ++c) sum += expf(logits[c] - mx);
        red[0] = mx;
        red[1] = logf(sum);
    }
    __syncthreads();
    if (t < N_CLASSES) out[g * N_CLASSES + t] = logits[t] - red[0] - red[1];
}

// ----------------------------------------------------------------- launch
extern "C" void kernel_launch(void* const* d_in, const int* in_sizes, int n_in,
                              void* d_out, int out_size, void* d_ws, size_t ws_size,
                              hipStream_t stream) {
    (void)in_sizes; (void)n_in; (void)out_size; (void)ws_size;
    const float* x        = (const float*)d_in[0];
    const int*   ei       = (const int*)d_in[1];
    const int*   batch    = (const int*)d_in[2];
    const float* conv1_wl = (const float*)d_in[4];
    const float* conv1_wr = (const float*)d_in[5];
    const float* conv1_b  = (const float*)d_in[6];
    const float* convs_wl = (const float*)d_in[7];
    const float* convs_wr = (const float*)d_in[8];
    const float* convs_b  = (const float*)d_in[9];
    const float* gate_w   = (const float*)d_in[10];
    const float* gate_b   = (const float*)d_in[11];
    const float* lin1_w   = (const float*)d_in[12];
    const float* lin1_b   = (const float*)d_in[13];
    const float* lin2_w   = (const float*)d_in[14];
    const float* lin2_b   = (const float*)d_in[15];
    float* out = (float*)d_out;

    // workspace carve-up (floats)
    float* p = (float*)d_ws;
    float*    invDeg  = p; p += 50048;
    float*    agg     = p; p += N_NODES * HIDDEN;
    float*    hA      = p; p += N_NODES * HIDDEN;
    float*    hB      = p; p += N_NODES * HIDDEN;
    __bf16*   wT      = (__bf16*)p; p += 49152;        // 6 * 128*128 bf16
    float*    gateBuf = p; p += 50048;
    float*    gBuf    = p; p += 50048;
    unsigned* mEnc    = (unsigned*)p; p += 512;
    float*    mVal    = p; p += 512;
    float*    den     = p; p += 512;
    float*    pooled  = p; p += N_GRAPHS * HIDDEN;
    float*    out1    = p; p += N_GRAPHS * HIDDEN;

    const int WMAT = HIDDEN * HIDDEN;                  // 16384

    // degrees
    k_zero<<<196, 256, 0, stream>>>(invDeg, N_NODES);
    k_deg<<<(N_EDGES + 255) / 256, 256, 0, stream>>>(ei, invDeg);
    k_invdeg<<<196, 256, 0, stream>>>(invDeg);

    // weight transpose+convert (bf16 [N][K])
    k_wprep<<<64, 256, 0, stream>>>(conv1_wl,         wT + 0 * WMAT);
    k_wprep<<<64, 256, 0, stream>>>(conv1_wr,         wT + 1 * WMAT);
    k_wprep<<<64, 256, 0, stream>>>(convs_wl,         wT + 2 * WMAT);
    k_wprep<<<64, 256, 0, stream>>>(convs_wl + WMAT,  wT + 3 * WMAT);
    k_wprep<<<64, 256, 0, stream>>>(convs_wr,         wT + 4 * WMAT);
    k_wprep<<<64, 256, 0, stream>>>(convs_wr + WMAT,  wT + 5 * WMAT);

    const int gemmBlocks    = (N_NODES + 63) / 64;            // 782
    const int scatterBlocks = (N_EDGES * 32) / 256;           // 100000
    const int aggZeroBlocks = (N_NODES * HIDDEN) / 256;       // 25000

    // layer 1
    k_zero<<<aggZeroBlocks, 256, 0, stream>>>(agg, N_NODES * HIDDEN);
    k_scatter<<<scatterBlocks, 256, 0, stream>>>(ei, x, agg);
    k_sage_gemm<<<gemmBlocks, 128, 0, stream>>>(agg, x, invDeg,
        wT + 0 * WMAT, wT + 1 * WMAT, conv1_b, hA);
    // layer 2
    k_zero<<<aggZeroBlocks, 256, 0, stream>>>(agg, N_NODES * HIDDEN);
    k_scatter<<<scatterBlocks, 256, 0, stream>>>(ei, hA, agg);
    k_sage_gemm<<<gemmBlocks, 128, 0, stream>>>(agg, hA, invDeg,
        wT + 2 * WMAT, wT + 4 * WMAT, convs_b, hB);
    // layer 3
    k_zero<<<aggZeroBlocks, 256, 0, stream>>>(agg, N_NODES * HIDDEN);
    k_scatter<<<scatterBlocks, 256, 0, stream>>>(ei, hB, agg);
    k_sage_gemm<<<gemmBlocks, 128, 0, stream>>>(agg, hB, invDeg,
        wT + 3 * WMAT, wT + 5 * WMAT, convs_b + HIDDEN, hA);

    // attention pooling (zero mEnc/mVal/den/pooled in one shot: contiguous)
    k_zero<<<(3 * 512 + N_GRAPHS * HIDDEN + 255) / 256, 256, 0, stream>>>(
        (float*)mEnc, 3 * 512 + N_GRAPHS * HIDDEN);
    k_gate<<<(N_NODES + 7) / 8, 256, 0, stream>>>(hA, gate_w, gate_b, batch,
                                                  gateBuf, mEnc);
    k_mdec<<<2, 256, 0, stream>>>(mEnc, mVal);
    k_exp<<<196, 256, 0, stream>>>(gateBuf, batch, mVal, gBuf, den);
    k_pool<<<(N_NODES * 32) / 256 + 1, 256, 0, stream>>>(hA, gBuf, den, batch,
                                                         pooled);

    // head
    k_lin1<<<N_GRAPHS, 128, 0, stream>>>(pooled, lin1_w, lin1_b, out1);
    k_lin2<<<N_GRAPHS, 32, 0, stream>>>(out1, lin2_w, lin2_b, out);
}
